// NormalizingFlow_77790447665266
// MI455X (gfx1250) — matrix operand order repaired and checked by reading, not compiled, not run on hardware
//
#include <hip/hip_runtime.h>
#include <hip/hip_bf16.h>

typedef __attribute__((ext_vector_type(16))) _Float16 v16h;
typedef __attribute__((ext_vector_type(8)))  float    v8f;

#define NROWS 1048576
#define NFLOW 8
#define ROWS_PER_BLOCK 128
#define NBLK (NROWS / ROWS_PER_BLOCK)   // 8192

// workspace layout (bytes):
//   [0   ..  64) : Lc (3x3 cholesky factor, row-major, 9 floats)
//   [64  .. 128) : S (global scalar sum)
//   [128 .. 128+NBLK*4)          : per-block partial sums
//   [128+NBLK*4 .. +65536)       : W2 repacked to f16 B-fragment layout
//                                  idx = (((f*2+kchunk)*4+nb)*32+lane)*16+i

// ---------------------------------------------------------------------------
// Prep: 3x3 cholesky of (tril(L)+1e-6 I)(...)^T, and repack W2 -> f16 B frags
// ---------------------------------------------------------------------------
__global__ void prep_kernel(const float* __restrict__ Ltri,
                            const float* __restrict__ W2,
                            float* __restrict__ ws_lc,
                            _Float16* __restrict__ ws_w2h) {
    int tid = blockIdx.x * blockDim.x + threadIdx.x;
    if (tid == 0) {
        float L[3][3];
        for (int r = 0; r < 3; ++r)
            for (int c = 0; c < 3; ++c)
                L[r][c] = (c <= r) ? Ltri[r * 3 + c] : 0.0f;
        for (int r = 0; r < 3; ++r) L[r][r] += 1e-6f;
        float cov[3][3];
        for (int r = 0; r < 3; ++r)
            for (int c = 0; c < 3; ++c)
                cov[r][c] = L[r][0]*L[c][0] + L[r][1]*L[c][1] + L[r][2]*L[c][2];
        float c00 = sqrtf(cov[0][0]);
        float c10 = cov[1][0] / c00;
        float c20 = cov[2][0] / c00;
        float c11 = sqrtf(cov[1][1] - c10 * c10);
        float c21 = (cov[2][1] - c20 * c10) / c11;
        float c22 = sqrtf(cov[2][2] - c20 * c20 - c21 * c21);
        ws_lc[0] = c00; ws_lc[1] = 0.f; ws_lc[2] = 0.f;
        ws_lc[3] = c10; ws_lc[4] = c11; ws_lc[5] = 0.f;
        ws_lc[6] = c20; ws_lc[7] = c21; ws_lc[8] = c22;
    }
    // repack W2[f][n][k] into B-fragment lane order:
    // B (32x16, f16): lane = (half<<4)|n16, element i covers k = 8*half + i (i<8), 16 + 8*half + (i-8) (i>=8)
    int total = NFLOW * 2 * 4 * 32 * 16;  // 32768
    for (int idx = tid; idx < total; idx += gridDim.x * blockDim.x) {
        int i    =  idx        & 15;
        int lane = (idx >> 4)  & 31;
        int nb   = (idx >> 9)  & 3;
        int kc   = (idx >> 11) & 1;
        int f    =  idx >> 12;
        int n = nb * 16 + (lane & 15);
        int k = kc * 32 + (lane >> 4) * 8 + i + ((i >= 8) ? 8 : 0);
        ws_w2h[idx] = (_Float16)W2[(f * 64 + n) * 64 + k];
    }
}

// ---------------------------------------------------------------------------
// Main: 8 coupling layers; 64x64 layer on v_wmma_f32_16x16x32_f16
// ---------------------------------------------------------------------------
__global__ __launch_bounds__(256)
void flow_kernel(const float* __restrict__ bm,
                 const float* __restrict__ W1g, const float* __restrict__ b1g,
                 const float* __restrict__ b2g,
                 const float* __restrict__ W3g, const float* __restrict__ b3g,
                 const float* __restrict__ eps,
                 const float* __restrict__ ws_lc,
                 const _Float16* __restrict__ ws_w2h,
                 float* __restrict__ out,
                 float* __restrict__ ws_part) {
    __shared__ float sW1[512], sb1[512], sb2[512], sb3[32], sW3[2048];
    __shared__ float sH[8][16 * 68];  // per-wave 16x64 tile, stride 68 (pad)
    __shared__ float sP[8][16][4];
    __shared__ float sRed[256];

    const int tid  = threadIdx.x;
    const int wave = tid >> 5;
    const int lane = tid & 31;
    const int m    = lane & 15;   // row within 16-row tile
    const int hlf  = lane >> 4;   // half-wave id (K/N halves of fragments)

    for (int i = tid; i < 512; i += 256) { sW1[i] = W1g[i]; sb1[i] = b1g[i]; sb2[i] = b2g[i]; }
    for (int i = tid; i < 2048; i += 256) sW3[i] = W3g[i];
    if (tid < 32) sb3[tid] = b3g[tid];
    __syncthreads();

    const float Lc00 = ws_lc[0], Lc10 = ws_lc[3], Lc11 = ws_lc[4];
    const float Lc20 = ws_lc[6], Lc21 = ws_lc[7], Lc22 = ws_lc[8];

    const int row = blockIdx.x * ROWS_PER_BLOCK + wave * 16 + m;
    const float e0 = eps[row * 3 + 0];
    const float e1 = eps[row * 3 + 1];
    const float e2 = eps[row * 3 + 2];
    const float z0 = bm[0] + Lc00 * e0;                       // pass-through comp
    float zA = bm[1] + Lc10 * e0 + Lc11 * e1;                 // transformed comp 1
    float zB = bm[2] + Lc20 * e0 + Lc21 * e1 + Lc22 * e2;     // transformed comp 2
    float ld = 0.0f;

    for (int f = 0; f < NFLOW; ++f) {
        // ---- layer 1 (rank-1): build A fragments directly in register layout
        // A 16x32 f16: lane<16 owns row m, K = {0..7,16..23}; lane>=16: +8
        v16h a0, a1;
        #pragma unroll
        for (int i = 0; i < 16; ++i) {
            int kk = i + ((i >= 8) ? 8 : 0) + hlf * 8;
            int j0 = kk;        // K chunk 0 -> hidden unit j0
            int j1 = 32 + kk;   // K chunk 1
            float h0 = fmaxf(z0 * sW1[f * 64 + j0] + sb1[f * 64 + j0], 0.0f);
            float h1 = fmaxf(z0 * sW1[f * 64 + j1] + sb1[f * 64 + j1], 0.0f);
            a0[i] = (_Float16)h0;
            a1[i] = (_Float16)h1;
        }

        // ---- layer 2 (64x64): 4 column blocks x 2 K-chunks of WMMA
        #pragma unroll
        for (int nb = 0; nb < 4; ++nb) {
            const v16h bf0 = *(const v16h*)(ws_w2h + ((((f * 2 + 0) * 4 + nb) * 32 + lane) << 4));
            const v16h bf1 = *(const v16h*)(ws_w2h + ((((f * 2 + 1) * 4 + nb) * 32 + lane) << 4));
            const float cinit = sb2[f * 64 + nb * 16 + m];   // bias folded into C
            v8f acc;
            #pragma unroll
            for (int v = 0; v < 8; ++v) acc[v] = cinit;
            acc = __builtin_amdgcn_wmma_f32_16x16x32_f16(false, a0, false, bf0,
                                                         (short)0, acc, false, false);
            acc = __builtin_amdgcn_wmma_f32_16x16x32_f16(false, a1, false, bf1,
                                                         (short)0, acc, false, false);
            // relu + stage D fragment to LDS (row = v + 8*hlf, col = nb*16 + m)
            #pragma unroll
            for (int v = 0; v < 8; ++v) {
                float hv = fmaxf(acc[v], 0.0f);
                sH[wave][(v + 8 * hlf) * 68 + nb * 16 + m] = hv;
            }
        }
        __syncthreads();

        // ---- layer 3 (64 -> 4): lane computes outputs {2*hlf, 2*hlf+1} of row m
        {
            const int o0 = 2 * hlf, o1 = 2 * hlf + 1;
            float p0 = sb3[f * 4 + o0];
            float p1 = sb3[f * 4 + o1];
            const float4* h4 = (const float4*)&sH[wave][m * 68];
            const float4* wa = (const float4*)&sW3[(f * 4 + o0) * 64];
            const float4* wb = (const float4*)&sW3[(f * 4 + o1) * 64];
            #pragma unroll
            for (int q = 0; q < 16; ++q) {
                float4 h = h4[q], a = wa[q], b = wb[q];
                p0 += h.x * a.x + h.y * a.y + h.z * a.z + h.w * a.w;
                p1 += h.x * b.x + h.y * b.y + h.z * b.z + h.w * b.w;
            }
            sP[wave][m][o0] = p0;
            sP[wave][m][o1] = p1;
        }
        __syncthreads();

        // ---- coupling update (replicated across both half-waves)
        {
            float s0 = tanhf(sP[wave][m][0]);
            float s1 = tanhf(sP[wave][m][1]);
            float t0 = sP[wave][m][2];
            float t1 = sP[wave][m][3];
            zA = zA * __expf(s0) + t0;
            zB = zB * __expf(s1) + t1;
            ld += s0 + s1;
        }
    }

    // ---- outputs (lanes of half 0 only; half 1 holds identical replicas)
    float rowsum = (hlf == 0) ? (zA + zB) : 0.0f;
    if (hlf == 0) {
        out[row * 3 + 0] = z0;
        out[row * 3 + 1] = __expf(zA);
        out[row * 3 + 2] = __expf(zB);
        out[3 * NROWS + row] = ld;   // finalize adds the global scalar S
    }

    // deterministic per-block partial of sum(z1 + z2)
    sRed[tid] = rowsum;
    __syncthreads();
    #pragma unroll
    for (int s = 128; s > 0; s >>= 1) {
        if (tid < s) sRed[tid] += sRed[tid + s];
        __syncthreads();
    }
    if (tid == 0) ws_part[blockIdx.x] = sRed[0];
}

// ---------------------------------------------------------------------------
// Deterministic fixed-order reduction of per-block partials -> scalar S
// ---------------------------------------------------------------------------
__global__ void reduce_kernel(const float* __restrict__ part, float* __restrict__ Sout) {
    __shared__ float red[256];
    float s = 0.0f;
    for (int i = threadIdx.x; i < NBLK; i += 256) s += part[i];
    red[threadIdx.x] = s;
    __syncthreads();
    for (int st = 128; st > 0; st >>= 1) {
        if (threadIdx.x < st) red[threadIdx.x] += red[threadIdx.x + st];
        __syncthreads();
    }
    if (threadIdx.x == 0) Sout[0] = red[0];
}

__global__ void finalize_kernel(float* __restrict__ out, const float* __restrict__ Sp) {
    int i = blockIdx.x * blockDim.x + threadIdx.x;
    if (i < NROWS) out[3 * NROWS + i] += Sp[0];
}

// ---------------------------------------------------------------------------
extern "C" void kernel_launch(void* const* d_in, const int* in_sizes, int n_in,
                              void* d_out, int out_size, void* d_ws, size_t ws_size,
                              hipStream_t stream) {
    const float* Ltri = (const float*)d_in[0];
    const float* bm   = (const float*)d_in[1];
    const float* W1   = (const float*)d_in[2];
    const float* b1   = (const float*)d_in[3];
    const float* W2   = (const float*)d_in[4];
    const float* b2   = (const float*)d_in[5];
    const float* W3   = (const float*)d_in[6];
    const float* b3   = (const float*)d_in[7];
    const float* eps  = (const float*)d_in[8];
    float* out = (float*)d_out;

    char* ws = (char*)d_ws;
    float*    ws_lc   = (float*)(ws + 0);
    float*    ws_S    = (float*)(ws + 64);
    float*    ws_part = (float*)(ws + 128);
    _Float16* ws_w2h  = (_Float16*)(ws + 128 + NBLK * 4);

    prep_kernel<<<128, 256, 0, stream>>>(Ltri, W2, ws_lc, ws_w2h);
    flow_kernel<<<NBLK, 256, 0, stream>>>(bm, W1, b1, b2, W3, b3, eps,
                                          ws_lc, ws_w2h, out, ws_part);
    reduce_kernel<<<1, 256, 0, stream>>>(ws_part, ws_S);
    finalize_kernel<<<(NROWS + 255) / 256, 256, 0, stream>>>(out, ws_S);
}
